// TypeAwareVisualTokenPruner_36301063586302
// MI455X (gfx1250) — matrix operand order repaired and checked by reading, compiled, not run on hardware
//
#include <hip/hip_runtime.h>
#include <hip/hip_bf16.h>
#include <math.h>

typedef __attribute__((ext_vector_type(16))) _Float16 v16h;
typedef __attribute__((ext_vector_type(8)))  _Float16 v8h;
typedef __attribute__((ext_vector_type(8)))  float    v8f;

// ---------------- problem constants ----------------
constexpr int B_ = 8, S_ = 4096, H_ = 2048, E_ = 512;
constexpr int T_TEXT = 512;
constexpr int V_ = S_ - T_TEXT;            // 3584
constexpr int KDIM = 2 * H_ + E_;          // 4608
constexpr int K_SEM = 512;                 // ceil(1280*0.40)
constexpr int K_SPA = 807;                 // ceil(1152*0.70)
constexpr int K_MIX = 634;                 // ceil(1152*0.55)
constexpr int K_VIS = K_SEM + K_SPA + K_MIX; // 1953
constexpr int K_OUT = T_TEXT + K_VIS;        // 2465
constexpr int M_TOT = B_ * V_;               // 28672

// ---------------- GEMM tiling ----------------
constexpr int MBLK = 32;            // tokens per workgroup (2 WMMA M-tiles)
constexpr int KC = 512;             // K chunk staged in LDS
constexpr int KCHUNKS = KDIM / KC;  // 9 (region boundaries 2048/4096 are chunk-aligned)
constexpr int LDA = KC + 8;         // padded LDS row stride (halfs) -> conflict-free b128 frag loads
constexpr int TPB_GEMM = 512;       // 16 wave32 waves
constexpr int NT_PER_WAVE = 8;      // 8 live n-tiles per wave (64 acc VGPRs)
constexpr int NPASS = 2;            // 16 waves: 2 M-tiles x 8 col-groups x 128 cols x 2 passes = 2048

// ---------------- workspace layout (bytes) ----------------
constexpr size_t OFF_SUMMARY = 0;                                   // B*H f32
constexpr size_t OFF_COUNTS  = OFF_SUMMARY + (size_t)B_ * H_ * 4;   // B f32 (padded)
constexpr size_t OFF_SCORES  = OFF_COUNTS + 256;                    // B*V f32
constexpr size_t OFF_KEEP    = OFF_SCORES + (size_t)B_ * V_ * 4;    // B*V i32
constexpr size_t OFF_MERGED  = OFF_KEEP + (size_t)B_ * V_ * 4;      // B*K_OUT i32
constexpr size_t OFF_WT      = ((OFF_MERGED + (size_t)B_ * K_OUT * 4) + 255) & ~(size_t)255; // N*K f16

// ---------------- output layout (float elements) ----------------
constexpr size_t O_HID = 0;                                   // (B,K_OUT,H)
constexpr size_t O_ATT = O_HID + (size_t)B_ * K_OUT * H_;     // (B,K_OUT)
constexpr size_t O_POS = O_ATT + (size_t)B_ * K_OUT;          // (3,B,K_OUT)
constexpr size_t O_VM  = O_POS + (size_t)3 * B_ * K_OUT;      // (B,K_OUT)
constexpr size_t O_VT  = O_VM + (size_t)B_ * K_OUT;           // (B,K_OUT)
constexpr size_t O_MG  = O_VT + (size_t)B_ * K_OUT;           // (B,K_OUT)

// ============ init: zero summary/keep, count text tokens per batch ============
__global__ void init_kernel(const int* __restrict__ attn, const int* __restrict__ vmask,
                            float* __restrict__ summary, int* __restrict__ keep,
                            float* __restrict__ counts) {
  const int b = blockIdx.x, tid = threadIdx.x;
  __shared__ int red[1024];
  for (int i = tid; i < H_; i += 1024) summary[b * H_ + i] = 0.0f;
  for (int i = tid; i < V_; i += 1024) keep[b * V_ + i] = 0;
  int c = 0;
  for (int s = tid; s < S_; s += 1024)
    c += (attn[b * S_ + s] > 0 && vmask[b * S_ + s] == 0) ? 1 : 0;
  red[tid] = c;
  __syncthreads();
  for (int off = 512; off > 0; off >>= 1) {
    if (tid < off) red[tid] += red[tid + off];
    __syncthreads();
  }
  if (tid == 0) counts[b] = (float)red[0];
}

// ============ masked sum of hidden over text tokens ============
__global__ void summary_acc_kernel(const float* __restrict__ hidden, const int* __restrict__ attn,
                                   const int* __restrict__ vmask, float* __restrict__ summary) {
  const int b = blockIdx.x;
  const int h = blockIdx.y * 256 + threadIdx.x;
  const int s0 = blockIdx.z * 512;
  float acc = 0.0f;
  for (int s = s0; s < s0 + 512; ++s) {
    if (attn[b * S_ + s] > 0 && vmask[b * S_ + s] == 0)
      acc += hidden[((size_t)b * S_ + s) * H_ + h];
  }
  atomicAdd(&summary[b * H_ + h], acc);
}

__global__ void summary_div_kernel(float* __restrict__ summary, const float* __restrict__ counts) {
  const int i = blockIdx.x * blockDim.x + threadIdx.x;
  summary[i] = summary[i] / counts[i / H_];
}

// ============ W1 (K x N, f32) -> Wt (N x K, f16) tiled transpose ============
__global__ void wcvt_kernel(const float* __restrict__ W1, _Float16* __restrict__ wt) {
  __shared__ float tile[32][33];
  const int k0 = blockIdx.x * 32;
  const int n0 = blockIdx.y * 32;
  for (int r = threadIdx.y; r < 32; r += 8)
    tile[r][threadIdx.x] = W1[(size_t)(k0 + r) * H_ + n0 + threadIdx.x];
  __syncthreads();
  for (int r = threadIdx.y; r < 32; r += 8)
    wt[(size_t)(n0 + r) * KDIM + k0 + threadIdx.x] = (_Float16)tile[threadIdx.x][r];
}

// ============ fused WMMA GEMM + GELU + W2 reduction -> scores ============
__global__ __launch_bounds__(TPB_GEMM)
void score_gemm_kernel(const float* __restrict__ hidden, const int* __restrict__ vtypes,
                       const float* __restrict__ temb, const float* __restrict__ summary,
                       const _Float16* __restrict__ wt, const float* __restrict__ b1,
                       const float* __restrict__ w2, const float* __restrict__ b2,
                       float* __restrict__ scores) {
  __shared__ _Float16 A_lds[MBLK * LDA];
  __shared__ float score_lds[MBLK];
  const int tid = threadIdx.x;
  const int lane = tid & 31;
  const int wave = tid >> 5;                // 0..15
  const int mt = wave & 1;                  // M-tile owned by this wave
  const int nbase = (wave >> 1) * 128;      // col-group within a pass (8 groups x 128)
  const int tile_base = blockIdx.x * MBLK;  // tiles never span batches (V % 32 == 0)

  if (tid < MBLK) score_lds[tid] = 0.0f;

  // staging coordinates: 32 rows x 16 threads/row, 32 cols per thread
  const int sr = tid >> 4;
  const int sc = (tid & 15) * 32;
  const int stok = tile_base + sr;
  const int sb = stok / V_;
  const int sv = stok - sb * V_;
  const int srow = T_TEXT + sv;
  int svt = vtypes[sb * S_ + srow];
  svt = svt < 0 ? 0 : (svt > 2 ? 2 : svt);

  // fragment coordinates (ISA 7.12.2 layouts)
  const int arow = mt * 16 + (lane & 15);
  const int akoff = (lane & 16) ? 8 : 0;    // A: interleaved K {0-7,16-23}/{8-15,24-31}
  const int bkoff = (lane & 16) ? 8 : 0;    // B: interleaved K {0-7,16-23}/{8-15,24-31}
  const int bn = (lane & 15);

  float part[8];
#pragma unroll
  for (int e = 0; e < 8; ++e) part[e] = 0.0f;

  const v8f vzero = {0.f, 0.f, 0.f, 0.f, 0.f, 0.f, 0.f, 0.f};

  for (int npass = 0; npass < NPASS; ++npass) {
    const int n0 = npass * 1024 + nbase;
    v8f acc[NT_PER_WAVE];
#pragma unroll
    for (int i = 0; i < NT_PER_WAVE; ++i) acc[i] = vzero;

    for (int kc = 0; kc < KCHUNKS; ++kc) {
      __syncthreads();
      {  // stage 32 x 512 A chunk (f32 -> f16) into LDS
        const float* src;
        if (kc < 4)      src = hidden + ((size_t)sb * S_ + srow) * H_ + kc * KC + sc;
        else if (kc < 8) src = summary + (size_t)sb * H_ + (kc - 4) * KC + sc;
        else             src = temb + (size_t)svt * E_ + sc;
        _Float16* dst = A_lds + sr * LDA + sc;
#pragma unroll
        for (int j = 0; j < 32; j += 4) {
          float4 f = *(const float4*)(src + j);
          dst[j + 0] = (_Float16)f.x; dst[j + 1] = (_Float16)f.y;
          dst[j + 2] = (_Float16)f.z; dst[j + 3] = (_Float16)f.w;
        }
      }
      __syncthreads();
#pragma unroll 2
      for (int kt = 0; kt < KC / 32; ++kt) {
        const int kb = kt * 32 + akoff;
        const _Float16* ap = A_lds + arow * LDA + kb;
        v8h alo = *(const v8h*)ap, ahi = *(const v8h*)(ap + 16);
        v16h af = __builtin_shufflevector(alo, ahi, 0,1,2,3,4,5,6,7,8,9,10,11,12,13,14,15);
        const size_t kg = (size_t)kc * KC + kt * 32 + bkoff;
#pragma unroll
        for (int nt = 0; nt < NT_PER_WAVE; ++nt) {
          const int nrow = n0 + nt * 16 + bn;
          const _Float16* bp = wt + (size_t)nrow * KDIM + kg;
          v8h blo = *(const v8h*)bp, bhi = *(const v8h*)(bp + 16);
          v16h bf = __builtin_shufflevector(blo, bhi, 0,1,2,3,4,5,6,7,8,9,10,11,12,13,14,15);
          acc[nt] = __builtin_amdgcn_wmma_f32_16x16x32_f16(false, af, false, bf,
                                                           (short)0, acc[nt], false, false);
        }
      }
    }

    // fold this pass: exact GELU then dot with W2 (frees accumulators)
#pragma unroll
    for (int nt = 0; nt < NT_PER_WAVE; ++nt) {
      const int n = n0 + nt * 16 + bn;
      const float bb = b1[n];
      const float ww = w2[n];
#pragma unroll
      for (int e = 0; e < 8; ++e) {
        float x = acc[nt][e] + bb;
        part[e] += 0.5f * x * (1.0f + erff(x * 0.70710678118654752f)) * ww;
      }
    }
  }

  // reduce 16 lanes per C-half, then cross-wave via LDS float atomics
  const int hi8 = (lane & 16) ? 8 : 0;  // high half-lanes hold C rows M=r+8
#pragma unroll
  for (int e = 0; e < 8; ++e) {
    float p = part[e];
    p += __shfl_xor(p, 1, 32); p += __shfl_xor(p, 2, 32);
    p += __shfl_xor(p, 4, 32); p += __shfl_xor(p, 8, 32);
    if ((lane & 15) == 0) atomicAdd(&score_lds[mt * 16 + e + hi8], p);
  }
  __syncthreads();
  if (tid < MBLK) scores[tile_base + tid] = score_lds[tid] + b2[0];
}

// ============ per-(batch,type) top-k via LDS bitonic sort ============
__global__ void topk_kernel(const float* __restrict__ scores, const int* __restrict__ vtypes,
                            int* __restrict__ keep) {
  const int b = blockIdx.x / 3;
  const int t = blockIdx.x % 3;
  const int tid = threadIdx.x;
  __shared__ unsigned long long keys[2048];
  __shared__ int cnt;
  if (tid == 0) cnt = 0;
  __syncthreads();
  for (int v = tid; v < V_; v += 1024) {
    if (vtypes[b * S_ + T_TEXT + v] == t) {
      unsigned u = __float_as_uint(scores[b * V_ + v]);
      u = (u & 0x80000000u) ? ~u : (u | 0x80000000u);  // order-preserving transform
      int slot = atomicAdd(&cnt, 1);
      keys[slot] = ((unsigned long long)u << 32) | (unsigned)(~v);  // ties -> lower index wins
    }
  }
  __syncthreads();
  const int n = cnt;
  for (int i = tid; i < 2048; i += 1024)
    if (i >= n) keys[i] = 0ull;
  __syncthreads();
  for (int k = 2; k <= 2048; k <<= 1) {
    for (int j = k >> 1; j > 0; j >>= 1) {
      for (int i = tid; i < 2048; i += 1024) {
        int ixj = i ^ j;
        if (ixj > i) {
          bool dir = ((i & k) == 0);  // descending overall
          unsigned long long a = keys[i], c = keys[ixj];
          if ((a < c) == dir) { keys[i] = c; keys[ixj] = a; }
        }
      }
      __syncthreads();
    }
  }
  const int kq = (t == 0) ? K_SEM : (t == 1) ? K_SPA : K_MIX;
  for (int i = tid; i < kq; i += 1024) {
    unsigned vv = ~(unsigned)(keys[i] & 0xFFFFFFFFull);
    keep[b * V_ + vv] = 1;
  }
}

// ============ keep-mask compaction -> sorted merged indices ============
__global__ void compact_kernel(const int* __restrict__ keep, int* __restrict__ merged) {
  const int b = blockIdx.x;
  const int tid = threadIdx.x;
  __shared__ int s[1024];
  for (int i = tid; i < T_TEXT; i += 1024) merged[b * K_OUT + i] = i;
  int running = 0;
  for (int c = 0; c < 4; ++c) {
    const int v = c * 1024 + tid;
    const int x = (v < V_) ? keep[b * V_ + v] : 0;
    s[tid] = x;
    __syncthreads();
    for (int off = 1; off < 1024; off <<= 1) {
      int val = s[tid];
      int add = (tid >= off) ? s[tid - off] : 0;
      __syncthreads();
      s[tid] = val + add;
      __syncthreads();
    }
    const int incl = s[tid];
    const int tot = s[1023];
    if (x) merged[b * K_OUT + T_TEXT + running + (incl - x)] = T_TEXT + v;
    __syncthreads();
    running += tot;
  }
}

// ============ gathers ============
__global__ void gather_hidden_kernel(const int* __restrict__ merged, const float* __restrict__ hidden,
                                     float* __restrict__ out) {
  const int b = blockIdx.x / K_OUT, i = blockIdx.x % K_OUT;
  const int row = merged[b * K_OUT + i];
  const float4* src = (const float4*)(hidden + ((size_t)b * S_ + row) * H_);
  float4* dst = (float4*)(out + O_HID + ((size_t)b * K_OUT + i) * H_);
  for (int j = threadIdx.x; j < H_ / 4; j += blockDim.x) dst[j] = src[j];
}

__global__ void gather_misc_kernel(const int* __restrict__ merged, const int* __restrict__ pos_ids,
                                   const int* __restrict__ vmask, const int* __restrict__ vtypes,
                                   float* __restrict__ out) {
  const int idx = blockIdx.x * blockDim.x + threadIdx.x;
  if (idx >= B_ * K_OUT) return;
  const int b = idx / K_OUT, i = idx % K_OUT;
  const int row = merged[idx];
  out[O_ATT + idx] = 1.0f;
  for (int r = 0; r < 3; ++r)
    out[O_POS + ((size_t)r * B_ + b) * K_OUT + i] = (float)pos_ids[((size_t)r * B_ + b) * S_ + row];
  out[O_VM + idx] = (float)vmask[b * S_ + row];
  out[O_VT + idx] = (float)vtypes[b * S_ + row];
  out[O_MG + idx] = (float)row;
}

extern "C" void kernel_launch(void* const* d_in, const int* in_sizes, int n_in,
                              void* d_out, int out_size, void* d_ws, size_t ws_size,
                              hipStream_t stream) {
  (void)in_sizes; (void)n_in; (void)out_size; (void)ws_size;
  const float* hidden = (const float*)d_in[0];
  const int* attn = (const int*)d_in[1];
  const int* pos = (const int*)d_in[2];
  const int* vmask = (const int*)d_in[3];
  const int* vtyp = (const int*)d_in[4];
  const float* temb = (const float*)d_in[5];
  const float* W1 = (const float*)d_in[6];
  const float* b1 = (const float*)d_in[7];
  const float* W2 = (const float*)d_in[8];
  const float* b2 = (const float*)d_in[9];
  float* out = (float*)d_out;
  char* ws = (char*)d_ws;
  float* summary = (float*)(ws + OFF_SUMMARY);
  float* counts = (float*)(ws + OFF_COUNTS);
  float* scores = (float*)(ws + OFF_SCORES);
  int* keep = (int*)(ws + OFF_KEEP);
  int* merged = (int*)(ws + OFF_MERGED);
  _Float16* wt = (_Float16*)(ws + OFF_WT);

  init_kernel<<<B_, 1024, 0, stream>>>(attn, vmask, summary, keep, counts);
  summary_acc_kernel<<<dim3(B_, H_ / 256, S_ / 512), 256, 0, stream>>>(hidden, attn, vmask, summary);
  summary_div_kernel<<<(B_ * H_) / 256, 256, 0, stream>>>(summary, counts);
  wcvt_kernel<<<dim3(KDIM / 32, H_ / 32), dim3(32, 8), 0, stream>>>(W1, wt);
  score_gemm_kernel<<<M_TOT / MBLK, TPB_GEMM, 0, stream>>>(hidden, vtyp, temb, summary, wt, b1, W2, b2, scores);
  topk_kernel<<<B_ * 3, 1024, 0, stream>>>(scores, vtyp, keep);
  compact_kernel<<<B_, 1024, 0, stream>>>(keep, merged);
  gather_hidden_kernel<<<B_ * K_OUT, 256, 0, stream>>>(merged, hidden, out);
  gather_misc_kernel<<<(B_ * K_OUT + 255) / 256, 256, 0, stream>>>(merged, pos, vmask, vtyp, out);
}
